// PNA_23502061043980
// MI455X (gfx1250) — matrix operand order repaired and checked by reading, not compiled, not run on hardware
//
#include <hip/hip_runtime.h>

#define NL 4
#define T_ 5
#define FIN 75
#define FOUT 15
#define NGRAPH 256
#define KPAD 256          // 225 K padded to 8 * 32
#define NPAD 384          // 375 N padded to 24 * 16
#define NT_N 24
#define AVG_DEG_LOG 2.8332133440562162f
#define BN_EPS 1e-5f
#define STD_EPS 1e-5f

typedef __attribute__((ext_vector_type(16))) __bf16 v16bf;
typedef __attribute__((ext_vector_type(8)))  float  v8f;
typedef __attribute__((ext_vector_type(8)))  int    v8i;

__device__ __forceinline__ unsigned short f2bf(float f) {
  unsigned int u = __float_as_uint(f);
  u += 0x7FFFu + ((u >> 16) & 1u);          // round-to-nearest-even
  return (unsigned short)(u >> 16);
}
// order-preserving float<->uint encoding for atomic min/max
__device__ __forceinline__ unsigned int encf(float f) {
  unsigned int u = __float_as_uint(f);
  return (u & 0x80000000u) ? ~u : (u | 0x80000000u);
}
__device__ __forceinline__ float decf(unsigned int u) {
  unsigned int v = (u & 0x80000000u) ? (u & 0x7FFFFFFFu) : ~u;
  return __uint_as_float(v);
}

// ---------------- prep kernels ----------------

// h0 = node_emb[x]   (h has row stride 80)
__global__ void k_prep_h0(const int* __restrict__ x, const float* __restrict__ node_emb,
                          float* __restrict__ h, int N) {
  int idx = blockIdx.x * blockDim.x + threadIdx.x;
  if (idx >= N * FIN) return;
  int n = idx / FIN, f = idx % FIN;
  h[n * 80 + f] = node_emb[x[n] * FIN + f];
}

// e_enc table: [NL][4][80]  (only 4 distinct edge_attr values)
__global__ void k_prep_etab(const float* __restrict__ edge_emb,
                            const float* __restrict__ eew, const float* __restrict__ eeb,
                            float* __restrict__ etab) {
  int idx = blockIdx.x * blockDim.x + threadIdx.x;
  if (idx >= NL * 4 * FIN) return;
  int l = idx / (4 * FIN);
  int a = (idx / FIN) % 4;
  int f = idx % FIN;
  float acc = eeb[l * FIN + f];
  for (int d = 0; d < 50; ++d)
    acc += edge_emb[a * 50 + d] * eew[(l * 50 + d) * FIN + f];
  etab[(l * 4 + a) * 80 + f] = acc;
}

// Pre-swizzle pre_w into bf16 B-fragment layout:
// wfrag[(((l*8+kt)*24+nt)*32+lane)*8 + p] packs W[K][n],W[K+1][n]
// with K = kt*32 + (lane>=16 ? 16:0) + 2p, n = nt*16 + (lane&15)
__global__ void k_prep_wfrag(const float* __restrict__ pre_w, unsigned int* __restrict__ wfrag) {
  int idx = blockIdx.x * blockDim.x + threadIdx.x;
  const int total = NL * 8 * NT_N * 32;
  if (idx >= total) return;
  int lane = idx & 31;
  int rest = idx >> 5;
  int nt = rest % NT_N;
  int kt = (rest / NT_N) % 8;
  int l  = rest / (NT_N * 8);
  int n = nt * 16 + (lane & 15);
  int kbase = kt * 32 + ((lane >= 16) ? 16 : 0);
  unsigned int* dst = wfrag + (size_t)idx * 8;
#pragma unroll
  for (int p = 0; p < 8; ++p) {
    int K0 = kbase + p * 2;
    float v0 = 0.f, v1 = 0.f;
    if (n < T_ * FIN) {
      int t = n / FIN, g = n % FIN;
      const float* W = pre_w + ((size_t)(l * T_ + t) * 225) * FIN;
      if (K0     < 225) v0 = W[(size_t)K0 * FIN + g];
      if (K0 + 1 < 225) v1 = W[(size_t)(K0 + 1) * FIN + g];
    }
    dst[p] = (unsigned int)f2bf(v0) | ((unsigned int)f2bf(v1) << 16);
  }
}

// ---------------- per-layer kernels ----------------

__global__ void k_init_agg(float* __restrict__ sum, float* __restrict__ sum2,
                           unsigned int* __restrict__ mne, unsigned int* __restrict__ mxe,
                           unsigned int* __restrict__ cnt, float* __restrict__ bns, int N) {
  int idx = blockIdx.x * blockDim.x + threadIdx.x;
  if (idx < N * NPAD) {
    sum[idx] = 0.f; sum2[idx] = 0.f;
    mne[idx] = 0xFFFFFFFFu; mxe[idx] = 0u;
  }
  if (idx < N) cnt[idx] = 0u;
  if (idx < 2 * FIN) bns[idx] = 0.f;
}

__global__ void k_cnt(const int* __restrict__ ei, unsigned int* __restrict__ cnt, int E) {
  int e = blockIdx.x * blockDim.x + threadIdx.x;
  if (e < E) atomicAdd(&cnt[ei[E + e]], 1u);
}

// Fused: gather [h[dst]|h[src]|e_enc] -> bf16 A tile -> WMMA vs pre_w -> +bias
// -> atomic segment sum/sum2/min/max keyed by dst.
// All 8 A fragments are preloaded (16x ds_load_b128, one wait), then the 24
// WMMAs issue back-to-back; the 24 B fragments are loop-invariant and live in
// VGPRs across the whole tile loop.
__global__ void __launch_bounds__(256)
k_edge_wmma(const float* __restrict__ h, const int* __restrict__ ei,
            const int* __restrict__ attr, const float* __restrict__ etab,
            const unsigned int* __restrict__ wfrag, const float* __restrict__ preb,
            float* __restrict__ sum, float* __restrict__ sum2,
            unsigned int* __restrict__ mne, unsigned int* __restrict__ mxe, int E) {
  __shared__ unsigned short ldsA[16 * KPAD];
  __shared__ int sdst[16], ssrc[16], sattr[16];
  const int tid  = threadIdx.x;
  const int lane = tid & 31;
  const int w    = tid >> 5;
  const int row  = lane & 15;
  const int lhi  = (lane >= 16) ? 8 : 0;   // A-fragment K sub-offset
  const int nTiles = E >> 4;
  const v8i* wf = reinterpret_cast<const v8i*>(wfrag);

  // B fragments: loop-invariant -> registers for the whole kernel
  v16bf bF[8][3];
#pragma unroll
  for (int kt = 0; kt < 8; ++kt)
#pragma unroll
    for (int j = 0; j < 3; ++j)
      bF[kt][j] = __builtin_bit_cast(v16bf, wf[(kt * NT_N + (w * 3 + j)) * 32 + lane]);

  for (int tile = blockIdx.x; tile < nTiles; tile += gridDim.x) {
    if (tid < 16) {
      int e = tile * 16 + tid;
      ssrc[tid]  = ei[e];
      sdst[tid]  = ei[E + e];
      sattr[tid] = attr[e];
    }
    __syncthreads();
    // build bf16 A tile [16][256]
    for (int idx = tid; idx < 16 * KPAD; idx += 256) {
      int r = idx >> 8, k = idx & (KPAD - 1);
      float v = 0.f;
      if (k < FIN)            v = h[sdst[r] * 80 + k];
      else if (k < 2 * FIN)   v = h[ssrc[r] * 80 + (k - FIN)];
      else if (k < 3 * FIN)   v = etab[sattr[r] * 80 + (k - 2 * FIN)];
      ldsA[idx] = f2bf(v);
    }
    __syncthreads();

    // preload ALL A fragments (contiguous k order: k = kt*32 + 16*(lane/16) + s, same as wfrag), one dscnt wait
    v16bf aF[8];
#pragma unroll
    for (int kt = 0; kt < 8; ++kt) {
      v8i araw;
#pragma unroll
      for (int p = 0; p < 8; ++p) {
        int k = kt * 32 + lhi * 2 + p * 2;
        araw[p] = *reinterpret_cast<const unsigned int*>(&ldsA[row * KPAD + k]);
      }
      aF[kt] = __builtin_bit_cast(v16bf, araw);
    }

    v8f acc[3];
#pragma unroll
    for (int j = 0; j < 3; ++j) acc[j] = (v8f){0.f,0.f,0.f,0.f,0.f,0.f,0.f,0.f};

    // 24 WMMAs back-to-back, no LDS waits interleaved
#pragma unroll
    for (int kt = 0; kt < 8; ++kt)
#pragma unroll
      for (int j = 0; j < 3; ++j)
        acc[j] = __builtin_amdgcn_wmma_f32_16x16x32_bf16(
            false, aF[kt], false, bF[kt][j], (short)0, acc[j], false, false);

    // epilogue: +bias, atomic segment reductions by dst
#pragma unroll
    for (int j = 0; j < 3; ++j) {
      int ncol = (w * 3 + j) * 16 + row;
      if (ncol < T_ * FIN) {
        float bias = preb[ncol];
#pragma unroll
        for (int r = 0; r < 8; ++r) {
          int m_row = r + lhi;                 // D layout: VGPR r -> M=r (+8 hi half)
          float m = acc[j][r] + bias;
          size_t base = (size_t)sdst[m_row] * NPAD + ncol;
          atomicAdd(&sum[base], m);
          atomicAdd(&sum2[base], m * m);
          unsigned int em = encf(m);
          atomicMax(&mxe[base], em);
          atomicMin(&mne[base], em);
        }
      }
    }
    __syncthreads();
  }
}

// Per-node: finish moments, build scalers, post GEMM (T x 900x15), lin (75x75),
// and accumulate BN statistics.
__global__ void __launch_bounds__(128)
k_node_post(const float* __restrict__ sum, const float* __restrict__ sum2,
            const unsigned int* __restrict__ mne, const unsigned int* __restrict__ mxe,
            const unsigned int* __restrict__ cnt, const float* __restrict__ qw,
            const float* __restrict__ qb, const float* __restrict__ lw,
            const float* __restrict__ lb, float* __restrict__ hpre,
            float* __restrict__ bns) {
  __shared__ float s_agg[T_ * 300];
  __shared__ float s_post[T_ * FOUT];
  __shared__ float s_sc[2];
  const int node = blockIdx.x;
  const int tid = threadIdx.x;
  const unsigned int c = cnt[node];
  const float cf = (float)c;
  const float deg = fmaxf(cf, 1.f);
  if (tid == 0) {
    float amp = logf(fmaxf(cf, 1.f) + 1.f) / AVG_DEG_LOG;
    s_sc[0] = amp; s_sc[1] = 1.f / amp;
  }
  for (int idx = tid; idx < T_ * FIN; idx += 128) {
    int t = idx / FIN, g = idx % FIN;
    size_t base = (size_t)node * NPAD + idx;
    float mean  = sum[base] / deg;
    float mean2 = sum2[base] / deg;
    float stdv  = sqrtf(fmaxf(mean2 - mean * mean, 0.f) + STD_EPS);
    float mn = (c > 0) ? decf(mne[base]) : 0.f;
    float mx = (c > 0) ? decf(mxe[base]) : 0.f;
    s_agg[t * 300 +       g] = mean;
    s_agg[t * 300 +  75 + g] = mn;
    s_agg[t * 300 + 150 + g] = mx;
    s_agg[t * 300 + 225 + g] = stdv;
  }
  __syncthreads();
  const float amp = s_sc[0], inv = s_sc[1];
  // post: out(t,900) @ qw(t,900,15); out = [agg, agg*amp, agg/amp]
  for (int o = tid; o < T_ * FOUT; o += 128) {
    int t = o / FOUT, g = o % FOUT;
    float acc = qb[o];
    const float* qt = qw + (size_t)t * 900 * FOUT;
    const float* ag = s_agg + t * 300;
    for (int f = 0; f < 300; ++f) {
      float a = ag[f];
      acc += a * (qt[(size_t)f * FOUT + g]
                  + amp * qt[(size_t)(f + 300) * FOUT + g]
                  + inv * qt[(size_t)(f + 600) * FOUT + g]);
    }
    s_post[o] = acc;
  }
  __syncthreads();
  // lin: post(75) @ lw(75x75) + lb  ; accumulate BN stats
  for (int o = tid; o < FIN; o += 128) {
    float acc = lb[o];
    for (int i = 0; i < FIN; ++i) acc += s_post[i] * lw[i * FIN + o];
    hpre[(size_t)node * 80 + o] = acc;
    atomicAdd(&bns[o], acc);
    atomicAdd(&bns[FIN + o], acc * acc);
  }
}

__global__ void k_bn_finalize(const float* __restrict__ bns, const float* __restrict__ gamma,
                              const float* __restrict__ beta, float* __restrict__ bnp, int N) {
  int o = blockIdx.x * blockDim.x + threadIdx.x;
  if (o >= FIN) return;
  float mu  = bns[o] / (float)N;
  float var = bns[FIN + o] / (float)N - mu * mu;
  float sc  = gamma[o] * rsqrtf(var + BN_EPS);
  bnp[o]       = sc;
  bnp[FIN + o] = beta[o] - mu * sc;
}

__global__ void k_bn_apply(const float* __restrict__ hpre, const float* __restrict__ bnp,
                           float* __restrict__ h, int N) {
  int idx = blockIdx.x * blockDim.x + threadIdx.x;
  if (idx >= N * FIN) return;
  int n = idx / FIN, o = idx % FIN;
  float v = hpre[(size_t)n * 80 + o] * bnp[o] + bnp[FIN + o];
  h[(size_t)n * 80 + o] = fmaxf(v, 0.f);
}

// ---------------- readout ----------------

__global__ void k_zero_g(float* __restrict__ g) {
  int idx = blockIdx.x * blockDim.x + threadIdx.x;
  if (idx < NGRAPH * FIN) g[idx] = 0.f;
}

__global__ void k_gsum(const float* __restrict__ h, const int* __restrict__ batch,
                       float* __restrict__ g, int N) {
  int idx = blockIdx.x * blockDim.x + threadIdx.x;
  if (idx >= N * FIN) return;
  int n = idx / FIN, f = idx % FIN;
  atomicAdd(&g[batch[n] * FIN + f], h[(size_t)n * 80 + f]);
}

__global__ void __launch_bounds__(128)
k_mlp(const float* __restrict__ g, const float* __restrict__ w1, const float* __restrict__ b1,
      const float* __restrict__ w2, const float* __restrict__ b2,
      const float* __restrict__ w3, const float* __restrict__ b3, float* __restrict__ out) {
  __shared__ float gv[FIN], h1[50], h2[25];
  const int gr = blockIdx.x, tid = threadIdx.x;
  for (int i = tid; i < FIN; i += 128) gv[i] = g[gr * FIN + i];
  __syncthreads();
  if (tid < 50) {
    float a = b1[tid];
    for (int i = 0; i < FIN; ++i) a += gv[i] * w1[i * 50 + tid];
    h1[tid] = fmaxf(a, 0.f);
  }
  __syncthreads();
  if (tid < 25) {
    float a = b2[tid];
    for (int i = 0; i < 50; ++i) a += h1[i] * w2[i * 25 + tid];
    h2[tid] = fmaxf(a, 0.f);
  }
  __syncthreads();
  if (tid == 0) {
    float a = b3[0];
    for (int i = 0; i < 25; ++i) a += h2[i] * w3[i];
    out[gr] = a;
  }
}

// ---------------- launch ----------------

extern "C" void kernel_launch(void* const* d_in, const int* in_sizes, int n_in,
                              void* d_out, int out_size, void* d_ws, size_t ws_size,
                              hipStream_t stream) {
  const int*   x          = (const int*)d_in[0];
  const int*   edge_index = (const int*)d_in[1];
  const int*   edge_attr  = (const int*)d_in[2];
  const int*   batch      = (const int*)d_in[3];
  const float* node_emb   = (const float*)d_in[4];
  const float* edge_emb   = (const float*)d_in[5];
  const float* eew        = (const float*)d_in[6];
  const float* eeb        = (const float*)d_in[7];
  const float* pre_w      = (const float*)d_in[8];
  const float* pre_b      = (const float*)d_in[9];
  const float* post_w     = (const float*)d_in[10];
  const float* post_b     = (const float*)d_in[11];
  const float* lin_w      = (const float*)d_in[12];
  const float* lin_b      = (const float*)d_in[13];
  const float* bn_gamma   = (const float*)d_in[14];
  const float* bn_beta    = (const float*)d_in[15];
  const float* mw1 = (const float*)d_in[16];
  const float* mb1 = (const float*)d_in[17];
  const float* mw2 = (const float*)d_in[18];
  const float* mb2 = (const float*)d_in[19];
  const float* mw3 = (const float*)d_in[20];
  const float* mb3 = (const float*)d_in[21];
  float* out = (float*)d_out;

  const int N = in_sizes[0];
  const int E = in_sizes[2];

  char* ws = (char*)d_ws;
  size_t off = 0;
  auto carve = [&](size_t bytes) { size_t r = off; off = (off + bytes + 255) & ~(size_t)255; return r; };
  float*        h     = (float*)(ws + carve((size_t)N * 80 * 4));
  float*        hpre  = (float*)(ws + carve((size_t)N * 80 * 4));
  float*        etab  = (float*)(ws + carve((size_t)NL * 4 * 80 * 4));
  unsigned int* wfrag = (unsigned int*)(ws + carve((size_t)NL * 8 * NT_N * 32 * 8 * 4));
  float*        sum   = (float*)(ws + carve((size_t)N * NPAD * 4));
  float*        sum2  = (float*)(ws + carve((size_t)N * NPAD * 4));
  unsigned int* mne   = (unsigned int*)(ws + carve((size_t)N * NPAD * 4));
  unsigned int* mxe   = (unsigned int*)(ws + carve((size_t)N * NPAD * 4));
  unsigned int* cnt   = (unsigned int*)(ws + carve((size_t)N * 4));
  float*        bns   = (float*)(ws + carve(1024));
  float*        bnp   = (float*)(ws + carve(1024));
  float*        g     = (float*)(ws + carve((size_t)NGRAPH * FIN * 4));
  (void)ws_size; (void)n_in; (void)out_size;

  // one-time prep (cheap, rerun every launch for determinism)
  k_prep_h0<<<(N * FIN + 255) / 256, 256, 0, stream>>>(x, node_emb, h, N);
  k_prep_etab<<<(NL * 4 * FIN + 255) / 256, 256, 0, stream>>>(edge_emb, eew, eeb, etab);
  k_prep_wfrag<<<(NL * 8 * NT_N * 32 + 255) / 256, 256, 0, stream>>>(pre_w, wfrag);

  for (int l = 0; l < NL; ++l) {
    k_init_agg<<<(N * NPAD + 255) / 256, 256, 0, stream>>>(sum, sum2, mne, mxe, cnt, bns, N);
    k_cnt<<<(E + 255) / 256, 256, 0, stream>>>(edge_index, cnt, E);
    k_edge_wmma<<<2048, 256, 0, stream>>>(
        h, edge_index, edge_attr,
        etab + (size_t)l * 4 * 80,
        wfrag + (size_t)l * 8 * NT_N * 32 * 8,
        pre_b + (size_t)l * T_ * FIN,
        sum, sum2, mne, mxe, E);
    k_node_post<<<N, 128, 0, stream>>>(
        sum, sum2, mne, mxe, cnt,
        post_w + (size_t)l * T_ * 900 * FOUT,
        post_b + (size_t)l * T_ * FOUT,
        lin_w + (size_t)l * FIN * FIN,
        lin_b + (size_t)l * FIN,
        hpre, bns);
    k_bn_finalize<<<1, 128, 0, stream>>>(bns, bn_gamma + l * FIN, bn_beta + l * FIN, bnp, N);
    k_bn_apply<<<(N * FIN + 255) / 256, 256, 0, stream>>>(hpre, bnp, h, N);
  }

  k_zero_g<<<(NGRAPH * FIN + 255) / 256, 256, 0, stream>>>(g);
  k_gsum<<<(N * FIN + 255) / 256, 256, 0, stream>>>(h, batch, g, N);
  k_mlp<<<NGRAPH, 128, 0, stream>>>(g, mw1, mb1, mw2, mb2, mw3, mb3, out);
}
